// InvariantPointAttention_21878563406213
// MI455X (gfx1250) — matrix operand order repaired and checked by reading, not compiled, not run on hardware
//
#include <hip/hip_runtime.h>
#include <hip/hip_bf16.h>
#include <math.h>
#include <stdint.h>

typedef __bf16 bf16_t;
typedef __attribute__((ext_vector_type(16))) __bf16 v16bf;
typedef __attribute__((ext_vector_type(8)))  __bf16 v8bf;
typedef __attribute__((ext_vector_type(4)))  __bf16 v4bf;
typedef __attribute__((ext_vector_type(8)))  float  v8f;

namespace cfg {
constexpr int B = 2, N = 1024, DIM = 512, H = 8, DH = 64, PD = 4;
constexpr int DQ = 96;                    // 64 scalar + 12 point + 1 bias + 19 pad
constexpr int DV = 80;                    // 64 scalar + 12 point + 1 ones + 3 pad
constexpr int FEAT = H * (DH + PD * 4);   // 640
constexpr float SCALAR_SCALE = 0.088388347648318447f;  // (2*DH)^-0.5
constexpr float POINT_SCALE  = 0.16666666666666666f;   // (2*PD*4.5)^-0.5
constexpr float EPS = 1e-8f;
}

__device__ inline v8f wmma_bf16(v16bf a, v16bf b, v8f c) {
  return __builtin_amdgcn_wmma_f32_16x16x32_bf16(false, a, false, b, (short)0, c,
                                                 false, false);
}
__device__ inline v16bf cat8(v8bf lo, v8bf hi) {
  return __builtin_shufflevector(lo, hi, 0, 1, 2, 3, 4, 5, 6, 7, 8, 9, 10, 11,
                                 12, 13, 14, 15);
}

// ---------------------------------------------------------------------------
// Tensor Data Mover: 1-D linear global->LDS stream with hardware LDS padding
// (32B data + 16B pad, matching the 48B lane stride of the fragment arrays).
// ---------------------------------------------------------------------------
#if __has_builtin(__builtin_amdgcn_tensor_load_to_lds) && \
    __has_builtin(__builtin_amdgcn_s_wait_tensorcnt)
#define IPA_HAVE_TDM 1
typedef __attribute__((ext_vector_type(4))) unsigned int u32x4;
typedef __attribute__((ext_vector_type(4))) int i32x4;
typedef __attribute__((ext_vector_type(8))) int i32x8;

__device__ inline void tdm_load_1d_padded(const void* gptr, unsigned lds_off,
                                          unsigned units8B) {
  const unsigned long long ga = (unsigned long long)(uintptr_t)gptr;
  u32x4 g0;
  g0[0] = 1u;                                   // count=1, user mode, no gather
  g0[1] = lds_off;                              // D# lds_addr
  g0[2] = (unsigned)ga;                         // global_addr[31:0]
  g0[3] = (unsigned)((ga >> 32) & 0x1FFFFFFu) | (2u << 30);  // addr[56:32]|type=2
  i32x8 g1;
  // data_size=8B(3), pad_enable, pad_interval=8 dwords(2), pad_amount=4 dwords(3)
  g1[0] = (3 << 16) | (1 << 20) | (2 << 22) | (3 << 25);
  g1[1] = (int)((units8B & 0xFFFFu) << 16);     // tensor_dim0[15:0]  (bits 63:48)
  g1[2] = (int)((units8B >> 16) & 0xFFFFu);     // tensor_dim0[31:16]; dim1=0
  g1[3] = (int)((units8B & 0xFFFFu) << 16);     // tile_dim0 (bits 127:112)
  g1[4] = 0;                                    // tile_dim1=0, tile_dim2=0
  g1[5] = (int)units8B;                         // tensor_dim0_stride[31:0]
  g1[6] = 0;
  g1[7] = 0;
  const i32x4 z4 = {0, 0, 0, 0};
#if __clang_major__ >= 23
  const i32x8 z8 = {0, 0, 0, 0, 0, 0, 0, 0};
  __builtin_amdgcn_tensor_load_to_lds(g0, g1, z4, z4, z8, 0);
#else
  __builtin_amdgcn_tensor_load_to_lds(g0, g1, z4, z4, 0);
#endif
}
#endif

// ---------------------------------------------------------------------------
// Generic C[M,Ncols] = A[M,K] @ W[K,Ncols] (+bias), f32 in/out, bf16 WMMA math.
// ---------------------------------------------------------------------------
template <bool ADD_BIAS>
__global__ __launch_bounds__(256) void gemm_bf16_kernel(
    const float* __restrict__ A, const float* __restrict__ W,
    const float* __restrict__ bias, float* __restrict__ C,
    int M, int K, int Ncols) {
  __shared__ __align__(16) bf16_t As[128][40];
  __shared__ __align__(16) bf16_t BsT[32][40];

  const int tid  = threadIdx.x;
  const int lane = tid & 31;
  const int wave = tid >> 5;
  const int half = lane >> 4;
  const int l16  = lane & 15;
  const int mbase = blockIdx.y * 128;
  const int nbase = blockIdx.x * 32;

  v8f acc0 = {0.f, 0.f, 0.f, 0.f, 0.f, 0.f, 0.f, 0.f};
  v8f acc1 = acc0;

  const int ar  = tid >> 1;
  const int ac  = (tid & 1) * 16;
  const int wc  = tid & 31;
  const int wr0 = (tid >> 5) << 2;

  for (int k0 = 0; k0 < K; k0 += 32) {
    if (k0 + 32 < K)
      __builtin_prefetch(A + (size_t)(mbase + ar) * K + (k0 + 32), 0, 1);
    {
      const float* src = A + (size_t)(mbase + ar) * K + k0 + ac;
#pragma unroll
      for (int i = 0; i < 16; i += 4) {
        const float4 v = *(const float4*)(src + i);
        As[ar][ac + i + 0] = (bf16_t)v.x;
        As[ar][ac + i + 1] = (bf16_t)v.y;
        As[ar][ac + i + 2] = (bf16_t)v.z;
        As[ar][ac + i + 3] = (bf16_t)v.w;
      }
    }
    {
      v4bf p;
#pragma unroll
      for (int i = 0; i < 4; ++i)
        p[i] = (bf16_t)W[(size_t)(k0 + wr0 + i) * Ncols + (nbase + wc)];
      *(v4bf*)&BsT[wc][wr0] = p;
    }
    __syncthreads();

    v16bf af;
    {
      const bf16_t* arow = &As[wave * 16 + l16][0];
      const int kk = half * 16;
      af = cat8(*(const v8bf*)(arow + kk), *(const v8bf*)(arow + kk + 8));
    }
#pragma unroll
    for (int nt = 0; nt < 2; ++nt) {
      const bf16_t* brow = &BsT[nt * 16 + l16][half * 16];
      const v16bf bfm = cat8(*(const v8bf*)brow, *(const v8bf*)(brow + 8));
      if (nt == 0) acc0 = wmma_bf16(af, bfm, acc0);
      else         acc1 = wmma_bf16(af, bfm, acc1);
    }
    __syncthreads();
  }

#pragma unroll
  for (int g = 0; g < 8; ++g) {
    const int row = mbase + wave * 16 + half * 8 + g;
    const int c0 = nbase + l16;
    float v0 = acc0[g], v1 = acc1[g];
    if (ADD_BIAS) { v0 += bias[c0]; v1 += bias[c0 + 16]; }
    C[(size_t)row * Ncols + c0]      = v0;
    C[(size_t)row * Ncols + c0 + 16] = v1;
  }
}

// ---------------------------------------------------------------------------
// Pack Q/K/V. Q stays row-major [bh][n][96]. K and V are written in global
// BLOCK-FRAGMENT-MAJOR order: per 64-j block the bytes are exactly the linear
// stream the attention kernel wants in LDS, so tile loads are 1-D memcpys
// (TDM-able). V channel 76 = 1.0 => PV WMMA accumulates the softmax
// denominator for free (flash "l" == ones-channel of O).
// K stream:  [jb][ck(3)][jt(4)][lane(32)][e(16)]   (12288 B / block)
// V stream:  [jb][vt(5)][jc(2)][lane(32)][e(16)]   (10240 B / block)
// ---------------------------------------------------------------------------
__global__ __launch_bounds__(256) void pack_qkv_kernel(
    const float* __restrict__ QSf, const float* __restrict__ KSf,
    const float* __restrict__ VSf, const float* __restrict__ QPf,
    const float* __restrict__ KPf, const float* __restrict__ VPf,
    const float* __restrict__ qc, const float* __restrict__ qv,
    const float* __restrict__ kc, const float* __restrict__ kv,
    const float* __restrict__ pweights,
    bf16_t* __restrict__ Qp, bf16_t* __restrict__ KpF, bf16_t* __restrict__ VpF) {
  using namespace cfg;
  const int idx = blockIdx.x * blockDim.x + threadIdx.x;
  if (idx >= B * N * H) return;
  const int h = idx % H;
  const int n = (idx / H) % N;
  const int b = idx / (H * N);

  const float pw = log1pf(__expf(pweights[h]));  // softplus
  const float qscale = POINT_SCALE * pw;
  const int bn = b * N + n;

  const float* qs = QSf + (size_t)bn * (H * DH) + h * DH;
  const float* ks = KSf + (size_t)bn * (H * DH) + h * DH;
  const float* vs = VSf + (size_t)bn * (H * DH) + h * DH;
  const float* qp = QPf + (size_t)bn * (H * PD * 3) + h * PD * 3;
  const float* kp = KPf + (size_t)bn * (H * PD * 3) + h * PD * 3;
  const float* vp = VPf + (size_t)bn * (H * PD * 3) + h * PD * 3;

  bf16_t* qd = Qp  + ((size_t)(b * H + h) * N + n) * DQ;
  bf16_t* kb = KpF + (size_t)(b * H + h) * N * DQ;   // stream base for this bh
  bf16_t* vb = VpF + (size_t)(b * H + h) * N * DV;

  // position-dependent parts of the fragment-stream offsets
  const int jb = n >> 6, jt = (n >> 4) & 3, jc = (n >> 5) & 1;
  const int colK = n & 15;                  // K: lane = half(chan)*16 + (n&15)
  const int halfV = (n >> 4) & 1, eV = n & 15;
  const size_t kpos = (size_t)(jb * 12 + jt) * 512 + colK * 16;     // + ck*2048 + halfC*256 + e
  const size_t vpos = (size_t)jb * 5120 + jc * 512 + halfV * 256 + eV;  // + vt*1024 + colV*16

  auto kput = [&](int c, float x) {
    kb[kpos + (c >> 5) * 2048 + ((c >> 4) & 1) * 256 + (c & 15)] = (bf16_t)x;
  };
  auto vput = [&](int v, float x) {
    vb[vpos + (v >> 4) * 1024 + (v & 15) * 16] = (bf16_t)x;
  };

#pragma unroll 8
  for (int d = 0; d < DH; ++d) {
    qd[d] = (bf16_t)(SCALAR_SCALE * qs[d]);
    kput(d, ks[d]);
    vput(d, vs[d]);
  }

  float Rq[9], tq[3], Rk[9], tk[3];
#pragma unroll
  for (int i = 0; i < 9; ++i) { Rq[i] = qv[(size_t)bn * 9 + i]; Rk[i] = kv[(size_t)bn * 9 + i]; }
#pragma unroll
  for (int i = 0; i < 3; ++i) { tq[i] = qc[(size_t)bn * 3 + i]; tk[i] = kc[(size_t)bn * 3 + i]; }

  float k2 = 0.f;
#pragma unroll
  for (int d = 0; d < PD; ++d) {
    const float q0 = qp[d * 3 + 0], q1 = qp[d * 3 + 1], q2 = qp[d * 3 + 2];
    const float k0 = kp[d * 3 + 0], k1 = kp[d * 3 + 1], k2v = kp[d * 3 + 2];
    const float v0 = vp[d * 3 + 0], v1 = vp[d * 3 + 1], v2 = vp[d * 3 + 2];
#pragma unroll
    for (int c = 0; c < 3; ++c) {
      const float qg = Rq[c * 3 + 0] * q0 + Rq[c * 3 + 1] * q1 + Rq[c * 3 + 2] * q2 + tq[c];
      const float kg = Rk[c * 3 + 0] * k0 + Rk[c * 3 + 1] * k1 + Rk[c * 3 + 2] * k2v + tk[c];
      const float vg = Rk[c * 3 + 0] * v0 + Rk[c * 3 + 1] * v1 + Rk[c * 3 + 2] * v2 + tk[c];
      qd[DH + d * 3 + c] = (bf16_t)(qscale * qg);
      kput(DH + d * 3 + c, kg);
      vput(DH + d * 3 + c, vg);
      k2 += kg * kg;
    }
  }
  qd[76] = (bf16_t)1.0f;
  kput(76, -0.5f * qscale * k2);
#pragma unroll
  for (int c = 77; c < DQ; ++c) { qd[c] = (bf16_t)0.f; kput(c, 0.f); }
  vput(76, 1.0f);                       // ones channel -> softmax denominator
  vput(77, 0.f); vput(78, 0.f); vput(79, 0.f);
}

// ---------------------------------------------------------------------------
// Flash attention per (b,h). K/V tiles arrive as linear streams (TDM with
// hardware LDS padding when available); all operand reads are ds_load_b128
// pairs; softmax denominator comes out of the PV WMMA's ones-channel.
// ---------------------------------------------------------------------------
__global__ __launch_bounds__(128) void attn_kernel(
    const bf16_t* __restrict__ Qp, const bf16_t* __restrict__ KpF,
    const bf16_t* __restrict__ VpF, float* __restrict__ O) {
  using namespace cfg;
  constexpr int BI = 64, BJ = 64;
  constexpr int KUNITS = 3 * 4 * 32;   // 32B units per K tile (12288 B)
  constexpr int VUNITS = 5 * 2 * 32;   // 32B units per V tile (10240 B)
  __shared__ __align__(16) bf16_t KsF[3][4][32][24];
  __shared__ __align__(16) bf16_t VsF[5][2][32][24];
  __shared__ __align__(16) bf16_t Ps[4][16][32];

  const int tid  = threadIdx.x;
  const int lane = tid & 31;
  const int wave = tid >> 5;
  const int half = lane >> 4;
  const int l16  = lane & 15;

  const int bh = blockIdx.y;
  const int i0 = blockIdx.x * BI;

  const bf16_t* Qb  = Qp  + (size_t)bh * N * DQ;
  const bf16_t* KbF = KpF + (size_t)bh * N * DQ;
  const bf16_t* VbF = VpF + (size_t)bh * N * DV;

  v16bf qf[3];
  {
    const bf16_t* qrow = Qb + (size_t)(i0 + wave * 16 + l16) * DQ;
#pragma unroll
    for (int ck = 0; ck < 3; ++ck) {
      const bf16_t* p = qrow + ck * 32 + half * 16;
      qf[ck] = cat8(*(const v8bf*)p, *(const v8bf*)(p + 8));
    }
  }

  const v8f vzero = {0.f, 0.f, 0.f, 0.f, 0.f, 0.f, 0.f, 0.f};
  float m[8];
  v8f o[5];
#pragma unroll
  for (int g = 0; g < 8; ++g) m[g] = -3.0e30f;
#pragma unroll
  for (int t = 0; t < 5; ++t) o[t] = vzero;

  for (int jbk = 0; jbk < N / BJ; ++jbk) {
    const bf16_t* kblk = KbF + (size_t)jbk * (KUNITS * 16);
    const bf16_t* vblk = VbF + (size_t)jbk * (VUNITS * 16);
#ifdef IPA_HAVE_TDM
    if (wave == 0) {
      tdm_load_1d_padded(kblk, (unsigned)(uintptr_t)&KsF[0][0][0][0], KUNITS * 4);
      tdm_load_1d_padded(vblk, (unsigned)(uintptr_t)&VsF[0][0][0][0], VUNITS * 4);
      __builtin_amdgcn_s_wait_tensorcnt((short)0);
    }
    __syncthreads();
#else
    for (int u = tid; u < KUNITS; u += 128) {           // linear 32B units
      const uint4* s = (const uint4*)(kblk + u * 16);
      uint4* d = (uint4*)(&KsF[0][0][0][0] + (size_t)u * 24);
      d[0] = s[0]; d[1] = s[1];
    }
    for (int u = tid; u < VUNITS; u += 128) {
      const uint4* s = (const uint4*)(vblk + u * 16);
      uint4* d = (uint4*)(&VsF[0][0][0][0] + (size_t)u * 24);
      d[0] = s[0]; d[1] = s[1];
    }
    __syncthreads();
#endif

    // 12 back-to-back QK WMMAs over the whole 64-j block
    v8f s[4];
#pragma unroll
    for (int st = 0; st < 4; ++st) s[st] = vzero;
#pragma unroll
    for (int ck = 0; ck < 3; ++ck)
#pragma unroll
      for (int st = 0; st < 4; ++st) {
        const bf16_t* p = &KsF[ck][st][lane][0];
        s[st] = wmma_bf16(qf[ck], cat8(*(const v8bf*)p, *(const v8bf*)(p + 8)), s[st]);
      }

    // one deferred flash update per 64 j (max only; sum comes from ones-chan)
    float alpha[8];
#pragma unroll
    for (int g = 0; g < 8; ++g) {
      float rm = fmaxf(fmaxf(s[0][g], s[1][g]), fmaxf(s[2][g], s[3][g]));
      rm = fmaxf(rm, __shfl_xor(rm, 1, 16));
      rm = fmaxf(rm, __shfl_xor(rm, 2, 16));
      rm = fmaxf(rm, __shfl_xor(rm, 4, 16));
      rm = fmaxf(rm, __shfl_xor(rm, 8, 16));
      const float mn = fmaxf(m[g], rm);
      alpha[g] = __expf(m[g] - mn);
      m[g] = mn;
      s[0][g] = __expf(s[0][g] - mn);
      s[1][g] = __expf(s[1][g] - mn);
      s[2][g] = __expf(s[2][g] - mn);
      s[3][g] = __expf(s[3][g] - mn);
    }
#pragma unroll
    for (int t = 0; t < 5; ++t)
#pragma unroll
      for (int g = 0; g < 8; ++g) o[t][g] *= alpha[g];

    // two 32-j chunks: P transpose through per-wave LDS, then PV WMMAs
#pragma unroll
    for (int jc = 0; jc < 2; ++jc) {
#pragma unroll
      for (int g = 0; g < 8; ++g) {
        Ps[wave][half * 8 + g][l16]      = (bf16_t)s[jc * 2 + 0][g];
        Ps[wave][half * 8 + g][16 + l16] = (bf16_t)s[jc * 2 + 1][g];
      }
      v16bf pf;
      {
        const bf16_t* prow = &Ps[wave][l16][0];
        const int kk = half * 16;
        pf = cat8(*(const v8bf*)(prow + kk), *(const v8bf*)(prow + kk + 8));
      }
#pragma unroll
      for (int t = 0; t < 5; ++t) {
        const bf16_t* pv = &VsF[t][jc][lane][0];
        o[t] = wmma_bf16(pf, cat8(*(const v8bf*)pv, *(const v8bf*)(pv + 8)), o[t]);
      }
    }
    __syncthreads();
  }

  // softmax denominator = ones-channel (v-tile 4, col 12); broadcast per row
  float* Ob = O + (size_t)bh * N * DV;
#pragma unroll
  for (int g = 0; g < 8; ++g) {
    const float lsum = __shfl(o[4][g], half * 16 + 12, 32);
    const float inv = 1.f / lsum;
    const int row = i0 + wave * 16 + half * 8 + g;
#pragma unroll
    for (int t = 0; t < 5; ++t)
      Ob[(size_t)row * DV + t * 16 + l16] = o[t][g] * inv;
  }
}

// ---------------------------------------------------------------------------
// Rotate attended points back to local frame, compute norms, concat feats.
// ---------------------------------------------------------------------------
__global__ __launch_bounds__(256) void finalize_kernel(
    const float* __restrict__ O, const float* __restrict__ qc,
    const float* __restrict__ qv, float* __restrict__ feats) {
  using namespace cfg;
  const int idx = blockIdx.x * blockDim.x + threadIdx.x;
  if (idx >= B * N * H) return;
  const int h = idx % H;
  const int n = (idx / H) % N;
  const int b = idx / (H * N);
  const int bn = b * N + n;
  const float* ob = O + ((size_t)(b * H + h) * N + n) * DV;
  float* f = feats + (size_t)bn * FEAT;

#pragma unroll 8
  for (int d = 0; d < DH; ++d) f[h * DH + d] = ob[d];

  float R[9], t[3];
#pragma unroll
  for (int i = 0; i < 9; ++i) R[i] = qv[(size_t)bn * 9 + i];
#pragma unroll
  for (int i = 0; i < 3; ++i) t[i] = qc[(size_t)bn * 3 + i];

#pragma unroll
  for (int d = 0; d < PD; ++d) {
    const float g0 = ob[DH + d * 3 + 0] - t[0];
    const float g1 = ob[DH + d * 3 + 1] - t[1];
    const float g2 = ob[DH + d * 3 + 2] - t[2];
    float n2 = EPS;
#pragma unroll
    for (int c = 0; c < 3; ++c) {
      const float loc = R[0 * 3 + c] * g0 + R[1 * 3 + c] * g1 + R[2 * 3 + c] * g2;  // R^T
      f[H * DH + h * (PD * 3) + d * 3 + c] = loc;
      n2 += loc * loc;
    }
    f[H * DH + H * PD * 3 + h * PD + d] = sqrtf(n2);
  }
}

// ---------------------------------------------------------------------------
extern "C" void kernel_launch(void* const* d_in, const int* in_sizes, int n_in,
                              void* d_out, int out_size, void* d_ws, size_t ws_size,
                              hipStream_t stream) {
  using namespace cfg;
  (void)in_sizes; (void)n_in; (void)out_size; (void)ws_size;

  const float* tgt  = (const float*)d_in[0];
  const float* mem  = (const float*)d_in[1];
  const float* qc   = (const float*)d_in[2];
  const float* qv   = (const float*)d_in[3];
  const float* kc   = (const float*)d_in[4];
  const float* kv   = (const float*)d_in[5];
  const float* Wq_s = (const float*)d_in[6];
  const float* Wk_s = (const float*)d_in[7];
  const float* Wv_s = (const float*)d_in[8];
  const float* Wq_p = (const float*)d_in[9];
  const float* Wk_p = (const float*)d_in[10];
  const float* Wv_p = (const float*)d_in[11];
  const float* pwts = (const float*)d_in[12];
  const float* Wo   = (const float*)d_in[13];
  const float* bo   = (const float*)d_in[14];
  float* out = (float*)d_out;

  const int M = B * N;  // 2048
  char* ws = (char*)d_ws;
  auto take = [&](size_t bytes) -> char* {
    char* p = ws;
    ws += (bytes + 255) & ~((size_t)255);
    return p;
  };
  float*  QSf  = (float*) take((size_t)M * (H * DH) * 4);
  float*  KSf  = (float*) take((size_t)M * (H * DH) * 4);
  float*  VSf  = (float*) take((size_t)M * (H * DH) * 4);
  float*  QPf  = (float*) take((size_t)M * (H * PD * 3) * 4);
  float*  KPf  = (float*) take((size_t)M * (H * PD * 3) * 4);
  float*  VPf  = (float*) take((size_t)M * (H * PD * 3) * 4);
  bf16_t* Qpk  = (bf16_t*)take((size_t)B * H * N * DQ * 2);
  bf16_t* KpkF = (bf16_t*)take((size_t)B * H * N * DQ * 2);
  bf16_t* VpkF = (bf16_t*)take((size_t)B * H * N * DV * 2);
  float*  Oacc = (float*) take((size_t)B * H * N * DV * 4);
  float*  feat = (float*) take((size_t)M * FEAT * 4);

  const dim3 blk256(256);
  const dim3 gS(H * DH / 32, M / 128);
  const dim3 gP(H * PD * 3 / 32, M / 128);

  gemm_bf16_kernel<false><<<gS, blk256, 0, stream>>>(tgt, Wq_s, nullptr, QSf, M, DIM, H * DH);
  gemm_bf16_kernel<false><<<gS, blk256, 0, stream>>>(mem, Wk_s, nullptr, KSf, M, DIM, H * DH);
  gemm_bf16_kernel<false><<<gS, blk256, 0, stream>>>(mem, Wv_s, nullptr, VSf, M, DIM, H * DH);
  gemm_bf16_kernel<false><<<gP, blk256, 0, stream>>>(tgt, Wq_p, nullptr, QPf, M, DIM, H * PD * 3);
  gemm_bf16_kernel<false><<<gP, blk256, 0, stream>>>(mem, Wk_p, nullptr, KPf, M, DIM, H * PD * 3);
  gemm_bf16_kernel<false><<<gP, blk256, 0, stream>>>(mem, Wv_p, nullptr, VPf, M, DIM, H * PD * 3);

  const int nbh = B * N * H;
  pack_qkv_kernel<<<dim3((nbh + 255) / 256), blk256, 0, stream>>>(
      QSf, KSf, VSf, QPf, KPf, VPf, qc, qv, kc, kv, pwts, Qpk, KpkF, VpkF);

  attn_kernel<<<dim3(N / 64, B * H), dim3(128), 0, stream>>>(Qpk, KpkF, VpkF, Oacc);

  finalize_kernel<<<dim3((nbh + 255) / 256), blk256, 0, stream>>>(Oacc, qc, qv, feat);

  gemm_bf16_kernel<true><<<dim3(DIM / 32, M / 128), blk256, 0, stream>>>(
      feat, Wo, bo, out, M, FEAT, DIM);
}
